// StyleEncoder_74586402062763
// MI455X (gfx1250) — matrix operand order repaired and checked by reading, not compiled, not run on hardware
//
#include <hip/hip_runtime.h>

#define B_ 8

typedef __attribute__((ext_vector_type(16))) __bf16 bf16x16;
typedef __attribute__((ext_vector_type(8)))  float  f32x8;

static inline int cdiv(int a, int b) { return (a + b - 1) / b; }
static inline int cdivz(size_t a, int b) { return (int)((a + (size_t)b - 1) / (size_t)b); }
static inline int imax(int a, int b) { return a > b ? a : b; }

// ---------------------------------------------------------------------------
// Elementwise / utility kernels
// ---------------------------------------------------------------------------
__global__ void k_transpose(const float* __restrict__ x, float* __restrict__ xt, int N) {
  int i = blockIdx.x * blockDim.x + threadIdx.x;
  int total = B_ * N * 3;
  if (i >= total) return;
  int d = i % 3;
  int n = (i / 3) % N;
  int b = i / (3 * N);
  xt[((size_t)b * 3 + d) * N + n] = x[i];
}

__global__ void k_zero(float* __restrict__ p, size_t n) {
  size_t i = blockIdx.x * (size_t)blockDim.x + threadIdx.x;
  if (i < n) p[i] = 0.f;
}

__global__ void k_add(const float* __restrict__ a, const float* __restrict__ b,
                      float* __restrict__ o, size_t n) {
  size_t i = blockIdx.x * (size_t)blockDim.x + threadIdx.x;
  if (i < n) o[i] = a[i] + b[i];
}

// Pad channel rows: dst (B,Cd,N) <- src (B,Cs,N), rows >= Cs zero-filled.
__global__ void k_pad_rows(const float* __restrict__ src, float* __restrict__ dst,
                           int Cs, int Cd, int N) {
  size_t i = blockIdx.x * (size_t)blockDim.x + threadIdx.x;
  size_t total = (size_t)B_ * Cd * N;
  if (i >= total) return;
  int n = (int)(i % N);
  int c = (int)((i / N) % Cd);
  int b = (int)(i / ((size_t)Cd * N));
  dst[i] = (c < Cs) ? src[((size_t)b * Cs + c) * N + n] : 0.f;
}

// Pad 3D grid with 1-voxel zero halo + zero channel tail:
// dst (B,Cd,(r+2)^3) <- src (B,Cs,r^3)
__global__ void k_pad3d(const float* __restrict__ src, float* __restrict__ dst,
                        int r, int Cs, int Cd) {
  int rp = r + 2;
  int rp3 = rp * rp * rp;
  size_t i = blockIdx.x * (size_t)blockDim.x + threadIdx.x;
  size_t total = (size_t)B_ * Cd * rp3;
  if (i >= total) return;
  int pos = (int)(i % rp3);
  int c = (int)((i / rp3) % Cd);
  int b = (int)(i / ((size_t)Cd * rp3));
  int pz = pos % rp, py = (pos / rp) % rp, px = pos / (rp * rp);
  float v = 0.f;
  if (c < Cs && px >= 1 && px <= r && py >= 1 && py <= r && pz >= 1 && pz <= r)
    v = src[((size_t)b * Cs + c) * ((size_t)r * r * r) +
            (size_t)(((px - 1) * r + (py - 1)) * r + (pz - 1))];
  dst[i] = v;
}

// ---------------------------------------------------------------------------
// Weight packing: f32 (Co x Ktot) -> bf16 A-fragment order, K padded to 32.
// Flat element i = (((mt*nkc + kc)*32 + lane)*16 + e), value = W[m, ka]
// with m = mt*16 + (lane&15), ka = kc*32 + (lane&16?16:0) + e.
// ---------------------------------------------------------------------------
__global__ void k_pack_w(const float* __restrict__ src, __bf16* __restrict__ dst,
                         int Ktot, int nkc, int total) {
  int i = blockIdx.x * blockDim.x + threadIdx.x;
  if (i >= total) return;
  int e = i & 15;
  int lane = (i >> 4) & 31;
  int kc = (i >> 9) % nkc;
  int mt = i / (512 * nkc);
  int ka = kc * 32 + ((lane & 16) ? 16 : 0) + e;
  int m = mt * 16 + (lane & 15);
  dst[i] = (__bf16)((ka < Ktot) ? src[(size_t)m * Ktot + ka] : 0.f);
}

// ---------------------------------------------------------------------------
// Voxelization (directly into padded grid, zero halo)
// ---------------------------------------------------------------------------
__global__ void k_coord_stats(const float* __restrict__ coords, float* __restrict__ meanb,
                              float* __restrict__ maxnb, int N) {
  __shared__ float r0[256], r1[256], r2[256];
  __shared__ float smean[3];
  int b = blockIdx.x, tid = threadIdx.x;
  const float* cx = coords + (size_t)b * 3 * N;
  const float* cy = cx + N;
  const float* cz = cy + N;
  float sx = 0.f, sy = 0.f, sz = 0.f;
  for (int n = tid; n < N; n += 256) { sx += cx[n]; sy += cy[n]; sz += cz[n]; }
  r0[tid] = sx; r1[tid] = sy; r2[tid] = sz;
  __syncthreads();
  for (int s = 128; s > 0; s >>= 1) {
    if (tid < s) { r0[tid] += r0[tid + s]; r1[tid] += r1[tid + s]; r2[tid] += r2[tid + s]; }
    __syncthreads();
  }
  if (tid == 0) {
    smean[0] = r0[0] / N; smean[1] = r1[0] / N; smean[2] = r2[0] / N;
    meanb[b * 3 + 0] = smean[0]; meanb[b * 3 + 1] = smean[1]; meanb[b * 3 + 2] = smean[2];
  }
  __syncthreads();
  float mx = smean[0], my = smean[1], mz = smean[2];
  float mn = 0.f;
  for (int n = tid; n < N; n += 256) {
    float dx = cx[n] - mx, dy = cy[n] - my, dz = cz[n] - mz;
    mn = fmaxf(mn, sqrtf(dx * dx + dy * dy + dz * dz));
  }
  r0[tid] = mn;
  __syncthreads();
  for (int s = 128; s > 0; s >>= 1) {
    if (tid < s) r0[tid] = fmaxf(r0[tid], r0[tid + s]);
    __syncthreads();
  }
  if (tid == 0) maxnb[b] = r0[0];
}

__global__ void k_voxelize_pad(const float* __restrict__ coords, const float* __restrict__ meanb,
                               const float* __restrict__ maxnb, const float* __restrict__ feats,
                               float* __restrict__ pg, float* __restrict__ cnt,
                               float* __restrict__ cn, int Ci, int Cg, int N, int r) {
  int i = blockIdx.x * blockDim.x + threadIdx.x;
  if (i >= B_ * N) return;
  int b = i / N, n = i % N;
  int rp = r + 2;
  int rp3 = rp * rp * rp;
  int r3 = r * r * r;
  float inv = 1.f / (maxnb[b] * 2.f);
  int vi[3];
  for (int d = 0; d < 3; ++d) {
    float c = (coords[((size_t)b * 3 + d) * N + n] - meanb[b * 3 + d]) * inv + 0.5f;
    c = fminf(fmaxf(c * (float)r, 0.f), (float)(r - 1));
    cn[((size_t)b * 3 + d) * N + n] = c;
    vi[d] = (int)rintf(c);
  }
  int flat = (vi[0] * r + vi[1]) * r + vi[2];
  int pidx = (((vi[0] + 1) * rp) + (vi[1] + 1)) * rp + (vi[2] + 1);
  atomicAdd(&cnt[(size_t)b * r3 + flat], 1.f);
  for (int c = 0; c < Ci; ++c)
    atomicAdd(&pg[((size_t)b * Cg + c) * rp3 + pidx], feats[((size_t)b * Ci + c) * N + n]);
}

__global__ void k_voxel_div_pad(float* __restrict__ pg, const float* __restrict__ cnt,
                                int Ci, int Cg, int r) {
  int rp = r + 2;
  int rp3 = rp * rp * rp;
  int r3 = r * r * r;
  size_t i = blockIdx.x * (size_t)blockDim.x + threadIdx.x;
  size_t total = (size_t)B_ * Ci * r3;
  if (i >= total) return;
  int pos = (int)(i % r3);
  int c = (int)((i / r3) % Ci);
  int b = (int)(i / ((size_t)Ci * r3));
  int pz = pos % r, py = (pos / r) % r, px = pos / (r * r);
  int pidx = (((px + 1) * rp) + (py + 1)) * rp + (pz + 1);
  pg[((size_t)b * Cg + c) * rp3 + pidx] /= fmaxf(cnt[(size_t)b * r3 + pos], 1.f);
}

// ---------------------------------------------------------------------------
// WMMA GEMM: Y[b] (Co x N) = W (Co x K) * X[b] (K x N) + bias
// Packed bf16 A fragments, straight-line K loop (no guards), f32 accumulate.
// X must have nkc*32 zero-padded channel rows; N must be a multiple of 16.
// ---------------------------------------------------------------------------
__global__ void k_gemm_wmma(const __bf16* __restrict__ Wp, const float* __restrict__ bias,
                            const float* __restrict__ X, float* __restrict__ Y,
                            int nkc, int N) {
  int lane = threadIdx.x;
  int nt = blockIdx.x, mt = blockIdx.y, b = blockIdx.z;
  int Co = gridDim.y * 16;
  const float* Xb = X + (size_t)b * (size_t)(nkc * 32) * N;
  float* Yb = Y + (size_t)b * Co * N;
  int n = nt * 16 + (lane & 15);
  int kboff = (lane & 16) ? 16 : 0;
  const bf16x16* wp = (const bf16x16*)Wp + (size_t)mt * nkc * 32 + lane;
  f32x8 acc = {};
  for (int kc = 0; kc < nkc; ++kc) {
    bf16x16 afr = wp[(size_t)kc * 32];
    const float* xc = Xb + (size_t)(kc * 32 + kboff) * N + n;
    __builtin_prefetch(xc + (size_t)32 * N, 0, 0);
    bf16x16 bfr;
#pragma unroll
    for (int e = 0; e < 16; ++e) bfr[e] = (__bf16)xc[(size_t)e * N];
    acc = __builtin_amdgcn_wmma_f32_16x16x32_bf16(false, afr, false, bfr,
                                                  (short)0, acc, false, false);
  }
#pragma unroll
  for (int i = 0; i < 8; ++i) {
    int mm = mt * 16 + i + ((lane & 16) ? 8 : 0);
    Yb[(size_t)mm * N + n] = acc[i] + bias[mm];
  }
}

// ---------------------------------------------------------------------------
// 3x3x3 SAME conv (Co = 32) as implicit GEMM over K = Ci*27, via bf16 WMMA.
// Input is a halo-padded grid (B, Cg, (r+2)^3) with zeroed halo and spare
// channels, so the im2col gather needs no bounds checks at all.
// ---------------------------------------------------------------------------
__global__ void k_conv3d_wmma(const float* __restrict__ pg, const __bf16* __restrict__ Wp,
                              const float* __restrict__ bias, float* __restrict__ out,
                              int r, int Cg, int nkc) {
  int lane = threadIdx.x;
  int nt = blockIdx.x, mt = blockIdx.y, b = blockIdx.z;
  int rp = r + 2;
  int rp2 = rp * rp;
  int rp3 = rp2 * rp;
  int r3 = r * r * r;
  int pos = nt * 16 + (lane & 15);
  int pz = pos % r, py = (pos / r) % r, px = pos / (r * r);
  const float* base = pg + (size_t)b * Cg * rp3 +
                      (size_t)(((px + 1) * rp + (py + 1)) * rp + (pz + 1));
  int kboff = (lane & 16) ? 16 : 0;
  const bf16x16* wp = (const bf16x16*)Wp + (size_t)mt * nkc * 32 + lane;
  f32x8 acc = {};
  for (int kc = 0; kc < nkc; ++kc) {
    bf16x16 afr = wp[(size_t)kc * 32];
    bf16x16 bfr;
#pragma unroll
    for (int e = 0; e < 16; ++e) {
      int k = kc * 32 + kboff + e;
      int ci = k / 27;
      int t = k - ci * 27;
      int off = (t / 9) * rp2 + ((t / 3) % 3) * rp + (t % 3) - (rp2 + rp + 1);
      const float* p = base + (size_t)ci * rp3;
      bfr[e] = (__bf16)p[off];
    }
    acc = __builtin_amdgcn_wmma_f32_16x16x32_bf16(false, afr, false, bfr,
                                                  (short)0, acc, false, false);
  }
  float* ob = out + (size_t)b * 32 * r3;
#pragma unroll
  for (int i = 0; i < 8; ++i) {
    int mm = mt * 16 + i + ((lane & 16) ? 8 : 0);
    ob[(size_t)mm * r3 + pos] = acc[i] + bias[mm];
  }
}

// ---------------------------------------------------------------------------
// GroupNorm (groups=8), optional fused swish; in-place. One block per (b,g).
// ---------------------------------------------------------------------------
__global__ void k_group_norm(float* __restrict__ x, const float* __restrict__ scale,
                             const float* __restrict__ bias, int C, int S, int do_swish) {
  __shared__ float rs[256], rq[256];
  int b = blockIdx.x / 8, g = blockIdx.x % 8;
  int cpg = C / 8;
  float* base = x + ((size_t)b * C + (size_t)g * cpg) * S;
  size_t n = (size_t)cpg * S;
  float s = 0.f, q = 0.f;
  for (size_t i = threadIdx.x; i < n; i += 256) { float v = base[i]; s += v; q += v * v; }
  rs[threadIdx.x] = s; rq[threadIdx.x] = q;
  __syncthreads();
  for (int t = 128; t > 0; t >>= 1) {
    if ((int)threadIdx.x < t) { rs[threadIdx.x] += rs[threadIdx.x + t]; rq[threadIdx.x] += rq[threadIdx.x + t]; }
    __syncthreads();
  }
  float fn = (float)n;
  float mean = rs[0] / fn;
  float var = rq[0] / fn - mean * mean;
  float rstd = rsqrtf(var + 1e-5f);
  for (size_t i = threadIdx.x; i < n; i += 256) {
    int c = g * cpg + (int)(i / S);
    float v = (base[i] - mean) * rstd * scale[c] + bias[c];
    if (do_swish) v = v / (1.f + __expf(-v));
    base[i] = v;
  }
}

// ---------------------------------------------------------------------------
// SE block
// ---------------------------------------------------------------------------
__global__ void k_avgpool(const float* __restrict__ x, float* __restrict__ pool, int S) {
  __shared__ float rs[256];
  int bc = blockIdx.x;
  const float* p = x + (size_t)bc * S;
  float s = 0.f;
  for (int i = threadIdx.x; i < S; i += 256) s += p[i];
  rs[threadIdx.x] = s;
  __syncthreads();
  for (int t = 128; t > 0; t >>= 1) {
    if ((int)threadIdx.x < t) rs[threadIdx.x] += rs[threadIdx.x + t];
    __syncthreads();
  }
  if (threadIdx.x == 0) pool[bc] = rs[0] / (float)S;
}

__global__ void k_se_fc(const float* __restrict__ pool, const float* __restrict__ w1,
                        const float* __restrict__ b1, const float* __restrict__ w2,
                        const float* __restrict__ b2, float* __restrict__ sc,
                        int C, int Cm) {
  __shared__ float sp[32], s1[8];
  int b = blockIdx.x, t = threadIdx.x;
  if (t < C) sp[t] = pool[b * C + t];
  __syncthreads();
  if (t < Cm) {
    float a = b1[t];
    for (int c = 0; c < C; ++c) a += w1[t * C + c] * sp[c];
    s1[t] = fmaxf(a, 0.f);
  }
  __syncthreads();
  if (t < C) {
    float a = b2[t];
    for (int c = 0; c < Cm; ++c) a += w2[t * Cm + c] * s1[c];
    sc[b * C + t] = 1.f / (1.f + __expf(-a));
  }
}

__global__ void k_scale(float* __restrict__ x, const float* __restrict__ sc,
                        int S, size_t total) {
  size_t i = blockIdx.x * (size_t)blockDim.x + threadIdx.x;
  if (i >= total) return;
  x[i] *= sc[i / S];
}

// ---------------------------------------------------------------------------
// Trilinear devoxelize (C = 32)
// ---------------------------------------------------------------------------
__global__ void k_devox(const float* __restrict__ h, const float* __restrict__ cn,
                        float* __restrict__ out, int r, int N) {
  int i = blockIdx.x * blockDim.x + threadIdx.x;
  if (i >= B_ * N) return;
  int b = i / N, n = i % N;
  int r3 = r * r * r;
  float cx = cn[((size_t)b * 3 + 0) * N + n];
  float cy = cn[((size_t)b * 3 + 1) * N + n];
  float cz = cn[((size_t)b * 3 + 2) * N + n];
  int lx = (int)floorf(cx), ly = (int)floorf(cy), lz = (int)floorf(cz);
  float fx = cx - lx, fy = cy - ly, fz = cz - lz;
  float acc[32];
  for (int c = 0; c < 32; ++c) acc[c] = 0.f;
  const float* hb = h + (size_t)b * 32 * r3;
  for (int dx = 0; dx < 2; ++dx)
    for (int dy = 0; dy < 2; ++dy)
      for (int dz = 0; dz < 2; ++dz) {
        int ix = min(max(lx + dx, 0), r - 1);
        int iy = min(max(ly + dy, 0), r - 1);
        int iz = min(max(lz + dz, 0), r - 1);
        float w = (dx ? fx : 1.f - fx) * (dy ? fy : 1.f - fy) * (dz ? fz : 1.f - fz);
        const float* hp = hb + (size_t)((ix * r + iy) * r + iz);
        for (int c = 0; c < 32; ++c) acc[c] += w * hp[(size_t)c * r3];
      }
  for (int c = 0; c < 32; ++c) out[((size_t)b * 32 + c) * N + n] = acc[c];
}

// ---------------------------------------------------------------------------
// Attention (C = 32): flash-style, fully on the matrix pipe. One wave per
// (b, 16-query tile); 32 keys per iteration.
//   score tiles: 2x v_wmma (QK^T, K = C = 32)
//   online softmax: per-row stats via LDS-staged score tile
//   P x V:        2x v_wmma (K = 32 keys, N = 16 channels each)
// The probability tile needs NO cross-lane traffic: both half-waves own the
// full 32 exponentiated scores of their row, matching A-fragment ownership.
// ---------------------------------------------------------------------------
__global__ void k_attention(const float* __restrict__ q, const float* __restrict__ kmat,
                            const float* __restrict__ v, float* __restrict__ h, int S) {
  __shared__ float ptile[16 * 32];
  __shared__ float carr[16];
  __shared__ float larr[16];
  int lane = threadIdx.x;
  int b = blockIdx.y;
  int i0 = blockIdx.x * 16;
  const float* qb = q + (size_t)b * 32 * S;
  const float* kb = kmat + (size_t)b * 32 * S;
  const float* vb = v + (size_t)b * 32 * S;
  const float scale = 0.1767766952966369f;  // 32^-0.5
  int row = lane & 15;
  int chalf = lane >> 4;
  int kaoff = chalf ? 8 : 0;
  int kboff = chalf ? 16 : 0;
  float mrow = -3.0e38f, lrow = 0.f;
  f32x8 accP0 = {}, accP1 = {};
  bf16x16 afr;  // Q^T tile: A[m][kc] = q[kc, i0+m]; constant over key chunks
#pragma unroll
  for (int e = 0; e < 16; ++e) {
    int kc = kboff + e;
    afr[e] = (__bf16)qb[(size_t)kc * S + i0 + row];
  }
  for (int j0 = 0; j0 < S; j0 += 32) {
    // --- two 16x16 score tiles via WMMA ---
    bf16x16 bf0, bf1;
#pragma unroll
    for (int e = 0; e < 16; ++e) {
      int kc = kboff + e;
      bf0[e] = (__bf16)kb[(size_t)kc * S + j0 + row];
      bf1[e] = (__bf16)kb[(size_t)kc * S + j0 + 16 + row];
    }
    __builtin_prefetch(kb + (size_t)kboff * S + j0 + 32 + row, 0, 0);
    f32x8 s0 = {}, s1 = {};
    s0 = __builtin_amdgcn_wmma_f32_16x16x32_bf16(false, afr, false, bf0,
                                                 (short)0, s0, false, false);
    s1 = __builtin_amdgcn_wmma_f32_16x16x32_bf16(false, afr, false, bf1,
                                                 (short)0, s1, false, false);
#pragma unroll
    for (int i = 0; i < 8; ++i) {
      ptile[(i + 8 * chalf) * 32 + row] = s0[i] * scale;
      ptile[(i + 8 * chalf) * 32 + 16 + row] = s1[i] * scale;
    }
    __syncthreads();
    // --- online softmax over this chunk's 32 scores (row = lane&15) ---
    float sv[32];
    float smax = mrow;
#pragma unroll
    for (int j = 0; j < 32; ++j) { sv[j] = ptile[row * 32 + j]; smax = fmaxf(smax, sv[j]); }
    float corr = __expf(mrow - smax);
    float psum = 0.f;
#pragma unroll
    for (int j = 0; j < 32; ++j) { sv[j] = __expf(sv[j] - smax); psum += sv[j]; }
    lrow = lrow * corr + psum;
    mrow = smax;
    if (lane < 16) carr[row] = corr;
    __syncthreads();
    // --- P tile directly as A fragment (no LDS round-trip) ---
    bf16x16 pfr;
#pragma unroll
    for (int e = 0; e < 16; ++e)
      pfr[e] = (__bf16)sv[kboff + e];
    // --- V chunk as B fragments: channels [0,16) and [16,32) ---
    bf16x16 v0, v1;
#pragma unroll
    for (int e = 0; e < 16; ++e) {
      int j = kboff + e;
      v0[e] = (__bf16)vb[(size_t)row * S + j0 + j];
      v1[e] = (__bf16)vb[(size_t)(16 + row) * S + j0 + j];
    }
    __builtin_prefetch(vb + (size_t)row * S + j0 + 32 + kboff, 0, 0);
    // --- rescale running accumulators by per-row corr, then accumulate ---
#pragma unroll
    for (int i = 0; i < 8; ++i) {
      float cf = carr[i + 8 * chalf];
      accP0[i] *= cf;
      accP1[i] *= cf;
    }
    accP0 = __builtin_amdgcn_wmma_f32_16x16x32_bf16(false, pfr, false, v0,
                                                    (short)0, accP0, false, false);
    accP1 = __builtin_amdgcn_wmma_f32_16x16x32_bf16(false, pfr, false, v1,
                                                    (short)0, accP1, false, false);
    __syncthreads();
  }
  if (lane < 16) larr[row] = lrow;
  __syncthreads();
  // D fragment: column = channel (lane&15), rows = queries -> contiguous in i.
#pragma unroll
  for (int i = 0; i < 8; ++i) {
    int M = i + 8 * chalf;
    float lr = larr[M];
    h[((size_t)b * 32 + row) * S + i0 + M] = accP0[i] / lr;
    h[((size_t)b * 32 + 16 + row) * S + i0 + M] = accP1[i] / lr;
  }
}

// ---------------------------------------------------------------------------
// Farthest point sampling. One block per batch; LDS distance array.
// ---------------------------------------------------------------------------
__global__ void k_fps(const float* __restrict__ coords, int* __restrict__ out_idx,
                      int N, int M) {
  __shared__ float d[2048];
  __shared__ float redv[256];
  __shared__ int redi[256];
  __shared__ int s_last;
  int b = blockIdx.x, tid = threadIdx.x;
  const float* cx = coords + (size_t)b * 3 * N;
  const float* cy = cx + N;
  const float* cz = cy + N;
  for (int n = tid; n < N; n += 256) d[n] = 1e10f;
  if (tid == 0) s_last = 0;
  __syncthreads();
  for (int t = 0; t < M; ++t) {
    int last = s_last;
    if (tid == 0) out_idx[(size_t)b * M + t] = last;
    float px = cx[last], py = cy[last], pz = cz[last];
    float bestv = -1.f; int besti = 0;
    for (int n = tid; n < N; n += 256) {
      float dx = cx[n] - px, dy = cy[n] - py, dz = cz[n] - pz;
      float dn = fminf(d[n], dx * dx + dy * dy + dz * dz);
      d[n] = dn;
      if (dn > bestv) { bestv = dn; besti = n; }
    }
    redv[tid] = bestv; redi[tid] = besti;
    __syncthreads();
    for (int s = 128; s > 0; s >>= 1) {
      if (tid < s && redv[tid + s] > redv[tid]) { redv[tid] = redv[tid + s]; redi[tid] = redi[tid + s]; }
      __syncthreads();
    }
    if (tid == 0) s_last = redi[0];
    __syncthreads();
  }
}

__global__ void k_gather_ctr(const float* __restrict__ coords, const int* __restrict__ idx,
                             float* __restrict__ ctr, int N, int M) {
  int i = blockIdx.x * blockDim.x + threadIdx.x;
  if (i >= B_ * M) return;
  int b = i / M, m = i % M;
  int n = idx[(size_t)b * M + m];
  for (int d = 0; d < 3; ++d)
    ctr[((size_t)b * 3 + d) * M + m] = coords[((size_t)b * 3 + d) * N + n];
}

// Ball query replicating reference top_k(-idx) semantics: first K in-radius
// indices ascending; pad with first hit (or 0 if none).
__global__ void k_ball_query(const float* __restrict__ ctr, const float* __restrict__ coords,
                             int* __restrict__ nidx, int M, int N, float radius, int K) {
  int i = blockIdx.x * blockDim.x + threadIdx.x;
  if (i >= B_ * M) return;
  int b = i / M, m = i % M;
  float cx = ctr[((size_t)b * 3 + 0) * M + m];
  float cy = ctr[((size_t)b * 3 + 1) * M + m];
  float cz = ctr[((size_t)b * 3 + 2) * M + m];
  const float* px = coords + (size_t)b * 3 * N;
  const float* py = px + N;
  const float* pz = py + N;
  float r2 = radius * radius;
  int cnt = 0, first = 0, have = 0;
  int* o = nidx + ((size_t)b * M + m) * K;
  for (int n = 0; n < N; ++n) {
    float dx = px[n] - cx, dy = py[n] - cy, dz = pz[n] - cz;
    if (dx * dx + dy * dy + dz * dz < r2) {
      if (!have) { first = n; have = 1; }
      if (cnt < K) o[cnt] = n;
      ++cnt;
    }
  }
  int fill = have ? first : 0;
  for (int s = (cnt < K ? cnt : K); s < K; ++s) o[s] = fill;
}

__global__ void k_build_groups(const float* __restrict__ feats, const float* __restrict__ coords,
                               const float* __restrict__ ctr, const int* __restrict__ nidx,
                               float* __restrict__ h, int C, int Ch, int M, int N, int K) {
  // h has Ch (padded, pre-zeroed) channel rows; we write the first C+3.
  int i = blockIdx.x * blockDim.x + threadIdx.x;
  if (i >= B_ * M * K) return;
  int k = i % K;
  int m = (i / K) % M;
  int b = i / (M * K);
  int n = nidx[((size_t)b * M + m) * K + k];
  int S = M * K;
  size_t sp = (size_t)m * K + k;
  for (int d = 0; d < 3; ++d)
    h[((size_t)b * Ch + d) * S + sp] =
        coords[((size_t)b * 3 + d) * N + n] - ctr[((size_t)b * 3 + d) * M + m];
  for (int c = 0; c < C; ++c)
    h[((size_t)b * Ch + 3 + c) * S + sp] = feats[((size_t)b * C + c) * N + n];
}

__global__ void k_maxk(const float* __restrict__ h, float* __restrict__ out,
                       int C, int M, int K) {
  int i = blockIdx.x * blockDim.x + threadIdx.x;
  if (i >= B_ * C * M) return;
  int m = i % M;
  int c = (i / M) % C;
  int b = i / (C * M);
  const float* p = h + ((size_t)b * C + c) * ((size_t)M * K) + (size_t)m * K;
  float mx = p[0];
  for (int k = 1; k < K; ++k) mx = fmaxf(mx, p[k]);
  out[((size_t)b * C + c) * M + m] = mx;
}

// ---------------------------------------------------------------------------
// Head
// ---------------------------------------------------------------------------
__global__ void k_meanvar(const float* __restrict__ f, float* __restrict__ m,
                          float* __restrict__ v, int C, int N) {
  int i = blockIdx.x * blockDim.x + threadIdx.x;
  if (i >= B_ * N) return;
  int b = i / N, n = i % N;
  float s = 0.f, q = 0.f;
  for (int c = 0; c < C; ++c) {
    float x = f[((size_t)b * C + c) * N + n];
    s += x; q += x * x;
  }
  m[(size_t)b * N + n] = s / C;
  v[(size_t)b * N + n] = (q - s * s / C) / (C - 1);
}

__global__ void k_head(const float* __restrict__ m, const float* __restrict__ v,
                       const float* __restrict__ mw, const float* __restrict__ mb,
                       const float* __restrict__ vw, const float* __restrict__ vb,
                       const float* __restrict__ pe, float* __restrict__ out,
                       int Z, int N) {
  int i = blockIdx.x * blockDim.x + threadIdx.x;
  if (i >= B_ * Z) return;
  int b = i / Z, o = i % Z;
  float a = mb[o] + vb[o] + pe[o];
  for (int n = 0; n < N; ++n)
    a += m[(size_t)b * N + n] * mw[(size_t)o * N + n] +
         v[(size_t)b * N + n] * vw[(size_t)o * N + n];
  out[(size_t)b * Z + o] = a;
}

// ---------------------------------------------------------------------------
// Host-side orchestration
// ---------------------------------------------------------------------------
struct Arena {
  char* base; size_t off;
  float* f(size_t n) { return (float*)bytes(n * sizeof(float)); }
  int* i32(size_t n) { return (int*)bytes(n * sizeof(int)); }
  void* bytes(size_t nb) { void* r = base + off; off += ((nb + 255) & ~(size_t)255); return r; }
};

struct PvP {
  const float *c1w, *c1b, *g1s, *g1b, *c2w, *c2b, *g2s, *g2b;
  const float *se1w, *se1b, *se2w, *se2b, *ptw, *ptb, *ptgs, *ptgb;
  const float *qw, *qb, *kw, *kb, *vw, *vb, *ow, *ob, *ags, *agb;
};
struct PvPk {
  const __bf16 *c1, *c2, *pt, *q, *k, *v, *o;
  int nk1, nk2, nkpt;
};
struct SaL { const float *w, *b, *gs, *gb; };
struct SaPk { const __bf16 *w1, *w2; int nk1, nk2; };

static const __bf16* packW(Arena& ar, hipStream_t st, const float* w, int Co, int Ktot,
                           int* nkc_out) {
  int nkc = cdiv(Ktot, 32);
  int total = (Co / 16) * nkc * 512;
  __bf16* dst = (__bf16*)ar.bytes((size_t)total * sizeof(__bf16));
  k_pack_w<<<cdiv(total, 256), 256, 0, st>>>(w, dst, Ktot, nkc, total);
  *nkc_out = nkc;
  return dst;
}

static void run_pvconv(hipStream_t st, Arena a, const float* feats, const float* ptX,
                       const float* coords, int Ci, int N, int r,
                       const PvP& p, const PvPk& pk, bool att, float* out) {
  int r3 = r * r * r;
  int rp = r + 2;
  int rp3 = rp * rp * rp;
  int Cg1 = Ci + 2;           // channels in padded grid for conv1 (K padding slack)
  int CgA = imax(Cg1, 34);    // allocation also reused as conv2's padded input
  float* meanb = a.f(B_ * 3);
  float* maxnb = a.f(B_);
  float* cn = a.f((size_t)B_ * 3 * N);
  float* pgrid = a.f((size_t)B_ * CgA * rp3);
  float* cnt = a.f((size_t)B_ * r3);
  float* h1 = a.f((size_t)B_ * 32 * r3);
  float* h2 = a.f((size_t)B_ * 32 * r3);
  float* pool = a.f(B_ * 32);
  float* sc = a.f(B_ * 32);
  float* vox = a.f((size_t)B_ * 32 * N);
  float* pt = a.f((size_t)B_ * 32 * N);

  k_coord_stats<<<B_, 256, 0, st>>>(coords, meanb, maxnb, N);
  k_zero<<<cdivz((size_t)B_ * Cg1 * rp3, 256), 256, 0, st>>>(pgrid, (size_t)B_ * Cg1 * rp3);
  k_zero<<<cdivz((size_t)B_ * r3, 256), 256, 0, st>>>(cnt, (size_t)B_ * r3);
  k_voxelize_pad<<<cdiv(B_ * N, 256), 256, 0, st>>>(coords, meanb, maxnb, feats,
                                                    pgrid, cnt, cn, Ci, Cg1, N, r);
  k_voxel_div_pad<<<cdivz((size_t)B_ * Ci * r3, 256), 256, 0, st>>>(pgrid, cnt, Ci, Cg1, r);

  k_conv3d_wmma<<<dim3(r3 / 16, 2, B_), 32, 0, st>>>(pgrid, pk.c1, p.c1b, h1, r, Cg1, pk.nk1);
  k_group_norm<<<B_ * 8, 256, 0, st>>>(h1, p.g1s, p.g1b, 32, r3, 1);
  // Re-halo h1 (34-channel view over the same pgrid allocation) for conv2.
  k_pad3d<<<cdivz((size_t)B_ * 34 * rp3, 256), 256, 0, st>>>(h1, pgrid, r, 32, 34);
  k_conv3d_wmma<<<dim3(r3 / 16, 2, B_), 32, 0, st>>>(pgrid, pk.c2, p.c2b, h2, r, 34, pk.nk2);

  if (att) {
    k_group_norm<<<B_ * 8, 256, 0, st>>>(h2, p.g2s, p.g2b, 32, r3, 0);
    float* q = a.f((size_t)B_ * 32 * r3);
    float* kk = a.f((size_t)B_ * 32 * r3);
    float* v = a.f((size_t)B_ * 32 * r3);
    float* hh = a.f((size_t)B_ * 32 * r3);
    float* ho = a.f((size_t)B_ * 32 * r3);
    k_gemm_wmma<<<dim3(r3 / 16, 2, B_), 32, 0, st>>>(pk.q, p.qb, h2, q, 1, r3);
    k_gemm_wmma<<<dim3(r3 / 16, 2, B_), 32, 0, st>>>(pk.k, p.kb, h2, kk, 1, r3);
    k_gemm_wmma<<<dim3(r3 / 16, 2, B_), 32, 0, st>>>(pk.v, p.vb, h2, v, 1, r3);
    k_attention<<<dim3(r3 / 16, B_), 32, 0, st>>>(q, kk, v, hh, r3);
    k_gemm_wmma<<<dim3(r3 / 16, 2, B_), 32, 0, st>>>(pk.o, p.ob, hh, ho, 1, r3);
    k_add<<<cdivz((size_t)B_ * 32 * r3, 256), 256, 0, st>>>(h2, ho, h2, (size_t)B_ * 32 * r3);
    k_group_norm<<<B_ * 8, 256, 0, st>>>(h2, p.ags, p.agb, 32, r3, 1);
  } else {
    k_group_norm<<<B_ * 8, 256, 0, st>>>(h2, p.g2s, p.g2b, 32, r3, 1);
  }

  k_avgpool<<<B_ * 32, 256, 0, st>>>(h2, pool, r3);
  k_se_fc<<<B_, 32, 0, st>>>(pool, p.se1w, p.se1b, p.se2w, p.se2b, sc, 32, 4);
  k_scale<<<cdivz((size_t)B_ * 32 * r3, 256), 256, 0, st>>>(h2, sc, r3, (size_t)B_ * 32 * r3);

  k_devox<<<cdiv(B_ * N, 256), 256, 0, st>>>(h2, cn, vox, r, N);

  k_gemm_wmma<<<dim3(N / 16, 2, B_), 32, 0, st>>>(pk.pt, p.ptb, ptX, pt, pk.nkpt, N);
  k_group_norm<<<B_ * 8, 256, 0, st>>>(pt, p.ptgs, p.ptgb, 32, N, 1);
  k_add<<<cdivz((size_t)B_ * 32 * N, 256), 256, 0, st>>>(vox, pt, out, (size_t)B_ * 32 * N);
}

static void run_sa(hipStream_t st, Arena a, const float* feats, const float* coords,
                   int C, int N, int M, float radius, const SaL* L, const SaPk& pk,
                   int C1, int C2, float* outF, float* outC) {
  const int K = 32;
  int S = M * K;
  int Ch = pk.nk1 * 32;  // padded channel rows for the grouped tensor (>= C+3)
  int* idx = a.i32((size_t)B_ * M);
  int* nidx = a.i32((size_t)B_ * M * K);
  float* h = a.f((size_t)B_ * Ch * S);
  float* t1 = a.f((size_t)B_ * C1 * S);
  float* t2 = a.f((size_t)B_ * C2 * S);

  k_fps<<<B_, 256, 0, st>>>(coords, idx, N, M);
  k_gather_ctr<<<cdiv(B_ * M, 256), 256, 0, st>>>(coords, idx, outC, N, M);
  k_ball_query<<<cdiv(B_ * M, 256), 256, 0, st>>>(outC, coords, nidx, M, N, radius, K);
  k_zero<<<cdivz((size_t)B_ * Ch * S, 256), 256, 0, st>>>(h, (size_t)B_ * Ch * S);
  k_build_groups<<<cdiv(B_ * M * K, 256), 256, 0, st>>>(feats, coords, outC, nidx, h, C, Ch, M, N, K);

  k_gemm_wmma<<<dim3(S / 16, C1 / 16, B_), 32, 0, st>>>(pk.w1, L[0].b, h, t1, pk.nk1, S);
  k_group_norm<<<B_ * 8, 256, 0, st>>>(t1, L[0].gs, L[0].gb, C1, S, 1);
  k_gemm_wmma<<<dim3(S / 16, C2 / 16, B_), 32, 0, st>>>(pk.w2, L[1].b, t1, t2, pk.nk2, S);
  k_group_norm<<<B_ * 8, 256, 0, st>>>(t2, L[1].gs, L[1].gb, C2, S, 1);

  k_maxk<<<cdiv(B_ * C2 * M, 256), 256, 0, st>>>(t2, outF, C2, M, K);
}

extern "C" void kernel_launch(void* const* d_in, const int* in_sizes, int n_in,
                              void* d_out, int out_size, void* d_ws, size_t ws_size,
                              hipStream_t stream) {
  (void)in_sizes; (void)n_in; (void)out_size; (void)ws_size;
  const float* x = (const float*)d_in[0];
  int pi = 1;
  auto P = [&]() { return (const float*)d_in[pi++]; };
  auto loadPv = [&](bool att) {
    PvP p{};
    p.c1w = P(); p.c1b = P(); p.g1s = P(); p.g1b = P();
    p.c2w = P(); p.c2b = P(); p.g2s = P(); p.g2b = P();
    p.se1w = P(); p.se1b = P(); p.se2w = P(); p.se2b = P();
    p.ptw = P(); p.ptb = P(); p.ptgs = P(); p.ptgb = P();
    if (att) {
      p.qw = P(); p.qb = P(); p.kw = P(); p.kb = P();
      p.vw = P(); p.vb = P(); p.ow = P(); p.ob = P();
      p.ags = P(); p.agb = P();
    }
    return p;
  };
  auto loadSa = [&](SaL* L) {
    for (int i = 0; i < 2; ++i) { L[i].w = P(); L[i].b = P(); L[i].gs = P(); L[i].gb = P(); }
  };

  PvP pv0 = loadPv(false);
  PvP pv1 = loadPv(false);
  PvP pv2 = loadPv(true);
  PvP pv3 = loadPv(false);
  SaL sa0[2]; loadSa(sa0);
  SaL sa1[2]; loadSa(sa1);
  SaL sa2[2]; loadSa(sa2);
  const float* mean_w = P(); const float* mean_b = P();
  const float* var_w = P(); const float* var_b = P();
  const float* pe = P();

  Arena ar{(char*)d_ws, 0};
  float* xt = ar.f((size_t)B_ * 3 * 2048);
  float* xpad = ar.f((size_t)B_ * 32 * 2048);  // pv0 point-branch input, K-padded
  float* f0 = ar.f((size_t)B_ * 64 * 2048);
  float* f1 = ar.f((size_t)B_ * 64 * 2048);
  float* c0 = ar.f((size_t)B_ * 3 * 2048);
  float* c1 = ar.f((size_t)B_ * 3 * 2048);

  // Pre-pack all weights into bf16 A-fragment layout (persistent region).
  PvPk k0{}, k1{}, k2{}, k3{};
  k0.c1 = packW(ar, stream, pv0.c1w, 32, 3 * 27, &k0.nk1);
  k0.c2 = packW(ar, stream, pv0.c2w, 32, 32 * 27, &k0.nk2);
  k0.pt = packW(ar, stream, pv0.ptw, 32, 3, &k0.nkpt);
  k1.c1 = packW(ar, stream, pv1.c1w, 32, 32 * 27, &k1.nk1);
  k1.c2 = packW(ar, stream, pv1.c2w, 32, 32 * 27, &k1.nk2);
  k1.pt = packW(ar, stream, pv1.ptw, 32, 32, &k1.nkpt);
  k2.c1 = packW(ar, stream, pv2.c1w, 32, 32 * 27, &k2.nk1);
  k2.c2 = packW(ar, stream, pv2.c2w, 32, 32 * 27, &k2.nk2);
  k2.pt = packW(ar, stream, pv2.ptw, 32, 32, &k2.nkpt);
  int dummy;
  k2.q = packW(ar, stream, pv2.qw, 32, 32, &dummy);
  k2.k = packW(ar, stream, pv2.kw, 32, 32, &dummy);
  k2.v = packW(ar, stream, pv2.vw, 32, 32, &dummy);
  k2.o = packW(ar, stream, pv2.ow, 32, 32, &dummy);
  k3.c1 = packW(ar, stream, pv3.c1w, 32, 64 * 27, &k3.nk1);
  k3.c2 = packW(ar, stream, pv3.c2w, 32, 32 * 27, &k3.nk2);
  k3.pt = packW(ar, stream, pv3.ptw, 32, 64, &k3.nkpt);
  SaPk s0{}, s1{}, s2{};
  s0.w1 = packW(ar, stream, sa0[0].w, 32, 35, &s0.nk1);
  s0.w2 = packW(ar, stream, sa0[1].w, 32, 32, &s0.nk2);
  s1.w1 = packW(ar, stream, sa1[0].w, 32, 35, &s1.nk1);
  s1.w2 = packW(ar, stream, sa1[1].w, 64, 32, &s1.nk2);
  s2.w1 = packW(ar, stream, sa2[0].w, 64, 35, &s2.nk1);
  s2.w2 = packW(ar, stream, sa2[1].w, 128, 64, &s2.nk2);

  Arena scratch = ar;  // transient region, reused per stage

  k_transpose<<<cdiv(B_ * 2048 * 3, 256), 256, 0, stream>>>(x, xt, 2048);
  k_pad_rows<<<cdivz((size_t)B_ * 32 * 2048, 256), 256, 0, stream>>>(xt, xpad, 3, 32, 2048);

  run_pvconv(stream, scratch, xt, xpad, xt, 3, 2048, 32, pv0, k0, false, f0);
  run_pvconv(stream, scratch, f0, f0, xt, 32, 2048, 32, pv1, k1, false, f1);
  run_sa(stream, scratch, f1, xt, 32, 2048, 1024, 0.1f, sa0, s0, 32, 32, f0, c0);
  run_pvconv(stream, scratch, f0, f0, c0, 32, 1024, 16, pv2, k2, true, f1);
  run_sa(stream, scratch, f1, c0, 32, 1024, 256, 0.2f, sa1, s1, 32, 64, f0, c1);
  run_pvconv(stream, scratch, f0, f0, c1, 64, 256, 8, pv3, k3, false, f1);
  run_sa(stream, scratch, f1, c1, 32, 256, 128, 0.4f, sa2, s2, 64, 128, f0, c0);

  Arena head = scratch;
  float* mbuf = head.f((size_t)B_ * 128);
  float* vbuf = head.f((size_t)B_ * 128);
  k_meanvar<<<cdiv(B_ * 128, 256), 256, 0, stream>>>(f0, mbuf, vbuf, 128, 128);
  k_head<<<cdiv(B_ * 128, 256), 256, 0, stream>>>(mbuf, vbuf, mean_w, mean_b, var_w, var_b,
                                                  pe, (float*)d_out, 128, 128);
}
